// CapsuleNet_69896297775383
// MI455X (gfx1250) — matrix operand order repaired and checked by reading, not compiled, not run on hardware
//
#include <hip/hip_runtime.h>

// ---------------------------------------------------------------------------
// CapsuleNet forward for gfx1250 (MI455X).
//  - conv1/conv2/decoder as WMMA f16 implicit GEMMs (v_wmma_f32_16x16x32_f16)
//  - weights pre-converted to f16 [N][K]; B tiles via global_load_async_to_lds
//  - conv kernels use 128-wide N tiles (8 WMMA / wave / k-step)
//  - conv2 double-buffered: async B + register-staged A gather behind WMMAs
//  - branch-free, vectorized im2col gathers (v4i offset-table loads)
//  - fused LDS-resident dynamic routing (u_hat recomputed, never materialized)
// ---------------------------------------------------------------------------

typedef __attribute__((ext_vector_type(16))) _Float16 v16h;
typedef __attribute__((ext_vector_type(8)))  _Float16 v8h;
typedef __attribute__((ext_vector_type(8)))  float    v8f;
typedef __attribute__((ext_vector_type(4)))  int      v4i;

#define TM 128
#define TN 64
#define TN2 128
#define TK 32

#ifndef __has_builtin
#define __has_builtin(x) 0
#endif
#if __has_builtin(__builtin_amdgcn_global_load_async_to_lds_b128)
#define HAVE_ASYNC_LDS 1
#else
#define HAVE_ASYNC_LDS 0
#endif

__device__ __forceinline__ void async_cp_b128(const void* g, void* l) {
#if HAVE_ASYNC_LDS
    void* gnc = (void*)g;   // strip const, then addrspace-cast
    __builtin_amdgcn_global_load_async_to_lds_b128(
        (__attribute__((address_space(1))) v4i*)gnc,
        (__attribute__((address_space(3))) v4i*)l, 0, 0);
#else
    *(uint4*)l = *(const uint4*)g;
#endif
}

__device__ __forceinline__ void async_wait0() {
#if HAVE_ASYNC_LDS
#if __has_builtin(__builtin_amdgcn_s_wait_asynccnt)
    __builtin_amdgcn_s_wait_asynccnt(0);
#else
    asm volatile("s_wait_asynccnt 0" ::: "memory");
#endif
#endif
}

// A fragment: 16x32 f16 (ISA 7.12.2). lane L: M=L&15; lanes 0-15 hold
// K 0-7 & 16-23, lanes 16-31 hold K 8-15 & 24-31. Two ds b128 loads.
__device__ __forceinline__ v16h load_a_frag(const _Float16* Als, int mBase, int lane) {
    int m = mBase + (lane & 15);
    int ksel = (lane >> 4) << 3;
    const _Float16* row = Als + m * TK;
    v8h lo = *(const v8h*)(row + ksel);
    v8h hi = *(const v8h*)(row + 16 + ksel);
    return __builtin_shufflevector(lo, hi, 0,1,2,3,4,5,6,7,8,9,10,11,12,13,14,15);
}

// B fragment: 32x16 f16; Bls stored [n][k]; each lane reads K 8h..8h+7 and 16+8h..16+8h+7, like A.
__device__ __forceinline__ v16h load_b_frag(const _Float16* Bls, int n0, int lane) {
    int n = n0 + (lane & 15);
    int kg = (lane >> 4) << 3;
    const _Float16* col = Bls + n * TK + kg;
    v8h lo = *(const v8h*)(col);
    v8h hi = *(const v8h*)(col + 16);
    return __builtin_shufflevector(lo, hi, 0,1,2,3,4,5,6,7,8,9,10,11,12,13,14,15);
}

// ---------------------------------------------------------------------------
// prep: f16 weight copies in [N][K] layout + im2col byte-offset tables
// ---------------------------------------------------------------------------
__global__ void prep(const float* __restrict__ c1w, const float* __restrict__ c2w,
                     const float* __restrict__ d1w, const float* __restrict__ d2w,
                     const float* __restrict__ d3w,
                     _Float16* __restrict__ wt1, _Float16* __restrict__ wt2,
                     _Float16* __restrict__ wd1, _Float16* __restrict__ wd2,
                     _Float16* __restrict__ wd3,
                     int* __restrict__ offT1, int* __restrict__ offT2)
{
    int idx = blockIdx.x * 256 + threadIdx.x;
    if (idx < 256 * 20736) wt2[idx] = (_Float16)c2w[idx];                    // [n][k]
    if (idx < 256 * 96) {
        int n = idx / 96, k = idx - n * 96;
        wt1[idx] = (k < 81) ? (_Float16)c1w[n * 81 + k] : (_Float16)0.f;
    }
    if (idx < 512 * 160) {                                                   // d1_w [160,512] -> [512][160]
        int n = idx / 160, k = idx - n * 160;
        wd1[idx] = (_Float16)d1w[k * 512 + n];
    }
    if (idx < 1024 * 512) {                                                  // d2_w [512,1024] -> [1024][512]
        int n = idx >> 9, k = idx & 511;
        wd2[idx] = (_Float16)d2w[k * 1024 + n];
    }
    if (idx < 784 * 1024) {                                                  // d3_w [1024,784] -> [784][1024]
        int n = idx >> 10, k = idx & 1023;
        wd3[idx] = (_Float16)d3w[k * 784 + n];
    }
    if (idx < 20736) {                                                       // conv2 gather byte offsets (f16)
        int c = idx / 81, r = idx - c * 81;
        int ky = r / 9, kx = r - ky * 9;
        offT2[idx] = (c * 400 + ky * 20 + kx) * 2;
    }
    if (idx < 96)                                                            // conv1 gather byte offsets (f32)
        offT1[idx] = (idx < 81) ? ((idx / 9) * 28 + (idx % 9)) * 4 : -1;
}

// ---------------------------------------------------------------------------
// conv1: M = 512*400, N = 256 (tile 128), K = 81 padded to 96. h1 f16, relu.
// ---------------------------------------------------------------------------
__global__ __launch_bounds__(256) void conv1_wmma(
    const float* __restrict__ x, const _Float16* __restrict__ wt1,
    const int* __restrict__ offT1, const float* __restrict__ bias,
    _Float16* __restrict__ h1)
{
    __shared__ __align__(16) _Float16 Als[TM * TK];
    __shared__ __align__(16) _Float16 Bls[TN2 * TK];
    int t = threadIdx.x, lane = t & 31, wv = t >> 5;
    int mTile = blockIdx.x * TM, nTile = blockIdx.y * TN2;

    int rowA = t >> 1, cbA = (t & 1) << 4;
    int gm = mTile + rowA;
    int bb = gm / 400, rr = gm - bb * 400;
    int oy = rr / 20, ox = rr - oy * 20;
    const char* xb = (const char*)x + ((size_t)bb * 784 + oy * 28 + ox) * 4;

    int nB = t >> 1, kpB = (t & 1) << 4;   // 2 async b128 per thread
    const _Float16* bSrc = wt1 + (nTile + nB) * 96 + kpB;

    v8f acc[8] = {};
    for (int k0 = 0; k0 < 96; k0 += TK) {
        // branch-free gather: clamp offset to 0, select 0 on the loaded value
        const v4i* tp4 = (const v4i*)(offT1 + k0 + cbA);
        v4i ot[4];
        #pragma unroll
        for (int q = 0; q < 4; ++q) ot[q] = tp4[q];
        _Float16 tmp[16];
        #pragma unroll
        for (int e = 0; e < 16; ++e) {
            int o = ot[e >> 2][e & 3];
            int oc = o & ~(o >> 31);               // o<0 -> 0 (in-bounds)
            float v = *(const float*)(xb + oc);
            tmp[e] = (_Float16)((o >= 0) ? v : 0.f);
        }
        *(v8h*)&Als[rowA * TK + cbA]     = *(const v8h*)tmp;
        *(v8h*)&Als[rowA * TK + cbA + 8] = *(const v8h*)(tmp + 8);
        async_cp_b128(bSrc + k0,     &Bls[nB * TK + kpB]);
        async_cp_b128(bSrc + k0 + 8, &Bls[nB * TK + kpB + 8]);
        async_wait0();
        __syncthreads();
        v16h a = load_a_frag(Als, wv << 4, lane);
        #pragma unroll
        for (int nc = 0; nc < 8; ++nc) {
            v16h bf = load_b_frag(Bls, nc << 4, lane);
            acc[nc] = __builtin_amdgcn_wmma_f32_16x16x32_f16(
                false, a, false, bf, (short)0, acc[nc], false, false);
        }
        __syncthreads();
    }
    int mBase = mTile + (wv << 4) + ((lane >> 4) << 3);
    #pragma unroll
    for (int nc = 0; nc < 8; ++nc) {
        int n = nTile + (nc << 4) + (lane & 15);
        float bv = bias[n];
        #pragma unroll
        for (int r = 0; r < 8; ++r) {
            int m = mBase + r;
            int b2 = m / 400, r2 = m - b2 * 400;
            float vv = fmaxf(acc[nc][r] + bv, 0.f);
            h1[((size_t)b2 * 256 + n) * 400 + r2] = (_Float16)vv;
        }
    }
}

// ---------------------------------------------------------------------------
// conv2 (stride 2): M = 18432, N = 256 (tile 128), K = 20736. Double-buffered:
// async B b128 + register-staged table gather for A issued before the WMMAs.
// ---------------------------------------------------------------------------
__global__ __launch_bounds__(256) void conv2_wmma(
    const _Float16* __restrict__ h1, const _Float16* __restrict__ wt2,
    const int* __restrict__ offT2, const float* __restrict__ bias,
    float* __restrict__ h2)
{
    __shared__ __align__(16) _Float16 Als[2][TM * TK];
    __shared__ __align__(16) _Float16 Bls[2][TN2 * TK];
    int t = threadIdx.x, lane = t & 31, wv = t >> 5;
    int mTile = blockIdx.x * TM, nTile = blockIdx.y * TN2;

    int rowA = t >> 1, cbA = (t & 1) << 4;
    int gm = mTile + rowA;
    int bb = gm / 36, p = gm - bb * 36;
    int oy = p / 6, ox = p - oy * 6;
    const char* aBase = (const char*)h1 + (size_t)bb * 204800 + (size_t)((40 * oy + 2 * ox) << 1);

    int nB = t >> 1, kpB = (t & 1) << 4;   // 2 async b128 per thread
    const _Float16* bSrc = wt2 + (size_t)(nTile + nB) * 20736 + kpB;

    v8f acc[8] = {};
    {   // prologue: fill buffer 0 (k0 = 0)
        const v4i* tp4 = (const v4i*)(offT2 + cbA);
        v4i ot[4];
        #pragma unroll
        for (int q = 0; q < 4; ++q) ot[q] = tp4[q];
        _Float16 tmp[16];
        #pragma unroll
        for (int e = 0; e < 16; ++e) tmp[e] = *(const _Float16*)(aBase + ot[e >> 2][e & 3]);
        *(v8h*)&Als[0][rowA * TK + cbA]     = *(const v8h*)tmp;
        *(v8h*)&Als[0][rowA * TK + cbA + 8] = *(const v8h*)(tmp + 8);
        async_cp_b128(bSrc,     &Bls[0][nB * TK + kpB]);
        async_cp_b128(bSrc + 8, &Bls[0][nB * TK + kpB + 8]);
    }
    async_wait0();
    __syncthreads();

    for (int ks = 0; ks < 648; ++ks) {
        int cur = ks & 1;
        int kn = (ks + 1) * TK;
        bool hasNext = (ks + 1) < 648;
        _Float16 tmp[16];
        if (hasNext) {
            async_cp_b128(bSrc + kn,     &Bls[cur ^ 1][nB * TK + kpB]);
            async_cp_b128(bSrc + kn + 8, &Bls[cur ^ 1][nB * TK + kpB + 8]);
            const v4i* tp4 = (const v4i*)(offT2 + kn + cbA);
            v4i ot[4];
            #pragma unroll
            for (int q = 0; q < 4; ++q) ot[q] = tp4[q];
            #pragma unroll
            for (int e = 0; e < 16; ++e) tmp[e] = *(const _Float16*)(aBase + ot[e >> 2][e & 3]);
            __builtin_prefetch(bSrc + kn + TK, 0, 1);
        }
        v16h a = load_a_frag(&Als[cur][0], wv << 4, lane);
        #pragma unroll
        for (int nc = 0; nc < 8; ++nc) {
            v16h bf = load_b_frag(&Bls[cur][0], nc << 4, lane);
            acc[nc] = __builtin_amdgcn_wmma_f32_16x16x32_f16(
                false, a, false, bf, (short)0, acc[nc], false, false);
        }
        if (hasNext) {
            *(v8h*)&Als[cur ^ 1][rowA * TK + cbA]     = *(const v8h*)tmp;
            *(v8h*)&Als[cur ^ 1][rowA * TK + cbA + 8] = *(const v8h*)(tmp + 8);
        }
        async_wait0();
        __syncthreads();
    }
    int mBase = mTile + (wv << 4) + ((lane >> 4) << 3);
    #pragma unroll
    for (int nc = 0; nc < 8; ++nc) {
        int n = nTile + (nc << 4) + (lane & 15);
        float bv = bias[n];
        #pragma unroll
        for (int r = 0; r < 8; ++r) {
            int m = mBase + r;
            int b2 = m / 36, p2 = m - b2 * 36;
            h2[((size_t)b2 * 256 + n) * 36 + p2] = acc[nc][r] + bv;
        }
    }
}

// ---------------------------------------------------------------------------
// squash: h2[b, c, p] -> u[b, i, n], n = c>>5, i = 36*(c&31)+p; squash over n
// ---------------------------------------------------------------------------
__global__ void squash_u(const float* __restrict__ h2, float* __restrict__ u)
{
    int idx = blockIdx.x * 256 + threadIdx.x;      // b*1152 + i
    if (idx >= 512 * 1152) return;
    int b = idx / 1152, i = idx - b * 1152;
    int q = i / 36, p = i - q * 36;
    float s[8], s2 = 0.f;
    #pragma unroll
    for (int n = 0; n < 8; ++n) {
        float v = h2[((size_t)b * 256 + n * 32 + q) * 36 + p];
        s[n] = v; s2 += v * v;
    }
    float scale = (s2 / (1.f + s2)) * rsqrtf(s2 + 1e-8f);
    #pragma unroll
    for (int n = 0; n < 8; ++n) u[(size_t)idx * 8 + n] = s[n] * scale;
}

// ---------------------------------------------------------------------------
// fused dynamic routing (one block per batch element, all state in LDS)
// ---------------------------------------------------------------------------
__global__ __launch_bounds__(256) void routing(
    const float* __restrict__ u, const float* __restrict__ W,
    float* __restrict__ caps_out, float* __restrict__ pred_out,
    _Float16* __restrict__ dec16)
{
    extern __shared__ float smem[];
    float* u_l  = smem;                 // 1152*8
    float* blog = u_l + 1152 * 8;       // 1152*10
    float* c_l  = blog + 1152 * 10;     // 1152*10
    float* s_l  = c_l + 1152 * 10;      // 160
    float* v_l  = s_l + 160;            // 160
    float* red  = v_l + 160;            // 256
    __shared__ int pred_s;

    int b = blockIdx.x, t = threadIdx.x;
    for (int idx = t; idx < 1152 * 8; idx += 256) u_l[idx] = u[(size_t)b * 9216 + idx];
    for (int idx = t; idx < 11520; idx += 256) blog[idx] = 0.f;
    __syncthreads();

    for (int it = 0; it < 3; ++it) {
        for (int j = 0; j < 10; ++j) {      // c = softmax over i (axis=1)
            float mx = -1e30f;
            for (int i = t; i < 1152; i += 256) mx = fmaxf(mx, blog[i * 10 + j]);
            red[t] = mx; __syncthreads();
            for (int off = 128; off; off >>= 1) {
                if (t < off) red[t] = fmaxf(red[t], red[t + off]);
                __syncthreads();
            }
            mx = red[0]; __syncthreads();
            float sm = 0.f;
            for (int i = t; i < 1152; i += 256) {
                float e = __expf(blog[i * 10 + j] - mx);
                c_l[i * 10 + j] = e; sm += e;
            }
            red[t] = sm; __syncthreads();
            for (int off = 128; off; off >>= 1) {
                if (t < off) red[t] += red[t + off];
                __syncthreads();
            }
            float inv = 1.f / red[0]; __syncthreads();
            for (int i = t; i < 1152; i += 256) c_l[i * 10 + j] *= inv;
        }
        __syncthreads();
        if (t < 160) {                      // s[j,m] = sum_i c[i,j]*(u[i].W[i,j,:,m])
            int j = t >> 4, m = t & 15;
            float accv = 0.f;
            for (int i = 0; i < 1152; ++i) {
                const float* up = u_l + i * 8;
                const float* Wp = W + (((size_t)i * 10 + j) * 8) * 16 + m;
                float uh = 0.f;
                #pragma unroll
                for (int n = 0; n < 8; ++n) uh += up[n] * Wp[n * 16];
                accv += c_l[i * 10 + j] * uh;
            }
            s_l[t] = accv;
        }
        __syncthreads();
        if (t < 10) {                       // v = squash(s)
            float s2 = 0.f;
            for (int m = 0; m < 16; ++m) { float xx = s_l[t * 16 + m]; s2 += xx * xx; }
            float scale = (s2 / (1.f + s2)) * rsqrtf(s2 + 1e-8f);
            for (int m = 0; m < 16; ++m) v_l[t * 16 + m] = s_l[t * 16 + m] * scale;
        }
        __syncthreads();
        if (it < 2) {                       // b_log += v . u_hat
            for (int p = t; p < 11520; p += 256) {
                int i = p / 10, j = p - i * 10;
                const float* up = u_l + i * 8;
                const float* Wp = W + (((size_t)i * 10 + j) * 8) * 16;
                float dot = 0.f;
                #pragma unroll
                for (int m = 0; m < 16; ++m) {
                    float uh = 0.f;
                    #pragma unroll
                    for (int n = 0; n < 8; ++n) uh += up[n] * Wp[n * 16 + m];
                    dot += v_l[j * 16 + m] * uh;
                }
                blog[p] += dot;
            }
            __syncthreads();
        }
    }
    if (t < 160) caps_out[(size_t)b * 160 + t] = v_l[t];
    if (t == 0) {
        int best = 0; float bn = -1.f;
        for (int j = 0; j < 10; ++j) {
            float s2 = 0.f;
            for (int m = 0; m < 16; ++m) { float xx = v_l[j * 16 + m]; s2 += xx * xx; }
            if (s2 > bn) { bn = s2; best = j; }
        }
        pred_s = best;
        pred_out[b] = (float)best;
    }
    __syncthreads();
    if (t < 160)
        dec16[(size_t)b * 160 + t] = (_Float16)(((t >> 4) == pred_s) ? v_l[t] : 0.f);
}

// ---------------------------------------------------------------------------
// decoder GEMM: A f16 [M][K], Bw f16 [N][K], bias f32. Fully async LDS fills.
// ACT: 1 relu, 2 sigmoid. OUTH: 1 -> f16 out [M][N], 0 -> f32 out.
// ---------------------------------------------------------------------------
template <int ACT, int OUTH>
__global__ __launch_bounds__(256) void gemm_f16(
    const _Float16* __restrict__ A, const _Float16* __restrict__ Bw,
    const float* __restrict__ bias, void* __restrict__ Cout,
    int M, int N, int K)
{
    __shared__ __align__(16) _Float16 Als[TM * TK];
    __shared__ __align__(16) _Float16 Bls[TN * TK];
    int t = threadIdx.x, lane = t & 31, wv = t >> 5;
    int mTile = blockIdx.x * TM, nTile = blockIdx.y * TN;

    int rowA = t >> 1, segA = (t & 1) << 4;
    int gm = mTile + rowA;
    int nB = t >> 2, kpB = (t & 3) << 3;
    int gn = nTile + nB;

    v8f acc[4] = {};
    for (int k0 = 0; k0 < K; k0 += TK) {
        if (gm < M) {
            const _Float16* src = A + (size_t)gm * K + k0 + segA;
            async_cp_b128(src,     &Als[rowA * TK + segA]);
            async_cp_b128(src + 8, &Als[rowA * TK + segA + 8]);
        } else {
            v8h z = {};
            *(v8h*)&Als[rowA * TK + segA]     = z;
            *(v8h*)&Als[rowA * TK + segA + 8] = z;
        }
        if (gn < N) {
            async_cp_b128(Bw + (size_t)gn * K + k0 + kpB, &Bls[nB * TK + kpB]);
        } else {
            v8h z = {};
            *(v8h*)&Bls[nB * TK + kpB] = z;
        }
        async_wait0();
        __syncthreads();
        v16h a = load_a_frag(Als, wv << 4, lane);
        #pragma unroll
        for (int nc = 0; nc < 4; ++nc) {
            v16h bf = load_b_frag(Bls, nc << 4, lane);
            acc[nc] = __builtin_amdgcn_wmma_f32_16x16x32_f16(
                false, a, false, bf, (short)0, acc[nc], false, false);
        }
        __syncthreads();
    }
    int mBase = mTile + (wv << 4) + ((lane >> 4) << 3);
    #pragma unroll
    for (int nc = 0; nc < 4; ++nc) {
        int n = nTile + (nc << 4) + (lane & 15);
        if (n >= N) continue;
        float bv = bias[n];
        #pragma unroll
        for (int r = 0; r < 8; ++r) {
            int m = mBase + r;
            if (m >= M) continue;
            float vv = acc[nc][r] + bv;
            if (ACT == 1) vv = fmaxf(vv, 0.f);
            else if (ACT == 2) vv = 1.f / (1.f + __expf(-vv));
            if (OUTH) ((_Float16*)Cout)[(size_t)m * N + n] = (_Float16)vv;
            else      ((float*)Cout)[(size_t)m * N + n] = vv;
        }
    }
}

// ---------------------------------------------------------------------------
// launcher
// ---------------------------------------------------------------------------
extern "C" void kernel_launch(void* const* d_in, const int* in_sizes, int n_in,
                              void* d_out, int out_size, void* d_ws, size_t ws_size,
                              hipStream_t stream)
{
    const float* x       = (const float*)d_in[0];
    const float* conv1_w = (const float*)d_in[1];
    const float* conv1_b = (const float*)d_in[2];
    const float* conv2_w = (const float*)d_in[3];
    const float* conv2_b = (const float*)d_in[4];
    const float* W       = (const float*)d_in[5];
    const float* d1_w    = (const float*)d_in[6];
    const float* d1_b    = (const float*)d_in[7];
    const float* d2_w    = (const float*)d_in[8];
    const float* d2_b    = (const float*)d_in[9];
    const float* d3_w    = (const float*)d_in[10];
    const float* d3_b    = (const float*)d_in[11];

    char* ws = (char*)d_ws;
    size_t off = 0;
    _Float16* h1   = (_Float16*)(ws + off); off += (size_t)512 * 256 * 400 * 2;  // 104,857,600
    float*    h2   = (float*)(ws + off);    off += (size_t)512 * 9216 * 4;       // 18,874,368
    float*    u    = (float*)(ws + off);    off += (size_t)512 * 9216 * 4;       // 18,874,368
    _Float16* wt2  = (_Float16*)(ws + off); off += (size_t)256 * 20736 * 2;      // 10,616,832
    _Float16* wt1  = (_Float16*)(ws + off); off += (size_t)256 * 96 * 2;
    _Float16* wd1  = (_Float16*)(ws + off); off += (size_t)512 * 160 * 2;
    _Float16* wd2  = (_Float16*)(ws + off); off += (size_t)1024 * 512 * 2;
    _Float16* wd3  = (_Float16*)(ws + off); off += (size_t)784 * 1024 * 2;
    int*      offT2 = (int*)(ws + off);     off += (size_t)20736 * 4;
    int*      offT1 = (int*)(ws + off);     off += 512;
    // dec16 / r1 / r2 overlay the h1 region (h1 dead once conv2 has run)
    _Float16* dec16 = (_Float16*)ws;                                   // 512*160
    _Float16* r1    = (_Float16*)(ws + (size_t)512 * 160 * 2);         // 512*512
    _Float16* r2    = (_Float16*)(ws + (size_t)512 * 160 * 2 + (size_t)512 * 512 * 2); // 512*1024

    float* out  = (float*)d_out;
    float* caps = out;                    // 512*10*16 = 81920
    float* rec  = out + 81920;            // 512*784   = 401408
    float* pred = out + 81920 + 401408;   // 512

    prep<<<20736, 256, 0, stream>>>(conv1_w, conv2_w, d1_w, d2_w, d3_w,
                                    wt1, wt2, wd1, wd2, wd3, offT1, offT2);
    conv1_wmma<<<dim3(1600, 2), 256, 0, stream>>>(x, wt1, offT1, conv1_b, h1);
    conv2_wmma<<<dim3(144, 2), 256, 0, stream>>>(h1, wt2, offT2, conv2_b, h2);
    squash_u<<<2304, 256, 0, stream>>>(h2, u);
    routing<<<512, 256, (size_t)(1152 * 8 + 1152 * 10 * 2 + 160 + 160 + 256) * 4, stream>>>(
        u, W, caps, pred, dec16);
    gemm_f16<1, 1><<<dim3(4, 8), 256, 0, stream>>>(dec16, wd1, d1_b, r1, 512, 512, 160);
    gemm_f16<1, 1><<<dim3(4, 16), 256, 0, stream>>>(r1, wd2, d2_b, r2, 512, 1024, 512);
    gemm_f16<2, 0><<<dim3(4, 13), 256, 0, stream>>>(r2, wd3, d3_b, rec, 512, 784, 1024);
}
